// LlamaAttention_27470610825680
// MI455X (gfx1250) — matrix operand order repaired and checked by reading, not compiled, not run on hardware
//
#include <hip/hip_runtime.h>
#include <hip/hip_bf16.h>
#include <math.h>

// ---------------------------------------------------------------------------
// LlamaAttention on MI455X (gfx1250, wave32, WMMA 16x16x32 bf16 + TDM).
//   B=4, S=2048, HID=576, H=9, D=64, KV=3 (GQA rep=3)
// Pipeline:
//   1) QKV projection GEMMs (f32 in -> bf16 out) via v_wmma_f32_16x16x32_bf16
//   2) RoPE on bf16 Q and K
//   3) Flash-attention: Q/K tiles staged via Tensor Data Mover
//      (tensor_load_to_lds + s_wait_tensorcnt), QK^T / PV in WMMA
//   4) Output projection (bf16 in -> f32 out) via WMMA
// ---------------------------------------------------------------------------

typedef __attribute__((ext_vector_type(16))) __bf16 v16bf;
typedef __attribute__((ext_vector_type(8)))  __bf16 v8bf;
typedef __attribute__((ext_vector_type(8)))  float  v8f;
typedef __attribute__((ext_vector_type(4)))  unsigned int u32x4;
typedef __attribute__((ext_vector_type(8)))  unsigned int u32x8;

static constexpr int BB  = 4;     // batch
static constexpr int SS  = 2048;  // sequence
static constexpr int HID = 576;
static constexpr int NH  = 9;     // query heads
static constexpr int NKV = 3;     // kv heads
static constexpr int DH  = 64;    // head dim

__device__ inline v8f vzero8() { v8f z = {0.f,0.f,0.f,0.f,0.f,0.f,0.f,0.f}; return z; }

// Build an A-fragment (16x32 bf16, M striped over lanes 0..15 / 16..31,
// K split lo/hi per lane half) from an LDS row holding 32 contiguous K values.
__device__ inline v16bf make_afrag(const __bf16* rowK32, int half /*0|1*/) {
  const v8bf lo = *(const v8bf*)(rowK32 + half * 16);
  const v8bf hi = *(const v8bf*)(rowK32 + half * 16 + 8);
  v16bf a;
#pragma unroll
  for (int i = 0; i < 8; ++i) { a[i] = lo[i]; a[i + 8] = hi[i]; }
  return a;
}

__device__ inline v8f wmma_bf16(v16bf a, v16bf b, v8f c) {
  return __builtin_amdgcn_wmma_f32_16x16x32_bf16(false, a, false, b, (short)0, c,
                                                 false, false);
}

// ---------------------------------------------------------------------------
// Tensor Data Mover: 2D bf16 tile (tileW x tileH elements, source row stride
// srcStride elements) from global memory into contiguous LDS rows.
// D# packed per CDNA5 ISA (group0 128b, group1 256b);
// VADDR2/VADDR3 omitted (<=2D tensor). Wave-level op: caller guards to 1 wave.
// ---------------------------------------------------------------------------
__device__ inline void tdm_load_2d(unsigned ldsAddr, const void* gsrc,
                                   unsigned tileW, unsigned tileH,
                                   unsigned srcStride) {
  unsigned long long ga = (unsigned long long)gsrc;
  u32x4 g0;
  g0[0] = 1u;                                        // count=1, user descriptor
  g0[1] = ldsAddr;                                   // lds_addr [63:32]
  g0[2] = (unsigned)(ga & 0xFFFFFFFFu);              // global_addr[31:0]
  g0[3] = (unsigned)((ga >> 32) & 0x01FFFFFFu)       // global_addr[56:32]
        | (2u << 30);                                // type=2 ("image")
  u32x8 g1;
  g1[0] = (1u << 16);                                // data_size=1 (2 bytes)
  g1[1] = (tileW & 0xFFFFu) << 16;                   // tensor_dim0[15:0]
  g1[2] = ((tileW >> 16) & 0xFFFFu)                  // tensor_dim0[31:16]
        | ((tileH & 0xFFFFu) << 16);                 // tensor_dim1[15:0]
  g1[3] = ((tileH >> 16) & 0xFFFFu)                  // tensor_dim1[31:16]
        | ((tileW & 0xFFFFu) << 16);                 // tile_dim0
  g1[4] = (tileH & 0xFFFFu);                         // tile_dim1 (tile_dim2=0)
  g1[5] = srcStride;                                 // tensor_dim0_stride[31:0]
  g1[6] = 0u;                                        // stride hi / dim1_stride lo
  g1[7] = 0u;
  asm volatile("tensor_load_to_lds %0, %1" :: "s"(g0), "s"(g1) : "memory");
}

__device__ inline void tdm_wait0() {
  __builtin_amdgcn_s_wait_tensorcnt(0);
}

// ---------------------------------------------------------------------------
// Generic tiled WMMA GEMM:  C[M,N] = A[M,K] @ W[K,N]
//   block = 256 threads = 8 waves; block tile = 128(M) x 64(N); K step = 32.
//   A is f32 or bf16 (template), W is f32, C is f32 or bf16 (template).
// ---------------------------------------------------------------------------
template <bool A_IS_F32, bool C_IS_F32>
__global__ __launch_bounds__(256)
void wmma_gemm(const void* __restrict__ Ap, const float* __restrict__ W,
               void* __restrict__ Cp, int M, int N, int K) {
  __shared__ __attribute__((aligned(64))) __bf16 ldsA[128][32];  // M x Kstep
  __shared__ __attribute__((aligned(64))) __bf16 ldsWt[64][32];  // N x Kstep

  const int t    = threadIdx.x;
  const int lane = t & 31;
  const int w    = t >> 5;        // wave id 0..7 -> M sub-tile
  const int l16  = lane & 15;
  const int half = lane >> 4;     // 0: lanes 0-15, 1: lanes 16-31

  const int mbase = blockIdx.x * 128;
  const int nbase = blockIdx.y * 64;

  v8f acc[4] = {vzero8(), vzero8(), vzero8(), vzero8()};

  const float*  Af32 = (const float*)Ap;
  const __bf16* Abf  = (const __bf16*)Ap;

  for (int kb = 0; kb < K; kb += 32) {
    // prefetch next weight tile into cache while this tile is consumed
    if (kb + 32 < K)
      __builtin_prefetch(W + (size_t)(kb + 32) * N + nbase + (t & 63), 0, 1);

    // cooperative load: A tile 128x32 -> bf16 LDS
#pragma unroll
    for (int i = 0; i < 16; ++i) {
      int idx = t + i * 256;                 // 4096 elements
      int r = idx >> 5, c = idx & 31;
      size_t g = (size_t)(mbase + r) * K + (kb + c);
      ldsA[r][c] = A_IS_F32 ? (__bf16)Af32[g] : Abf[g];
    }
    // cooperative load: W tile 32x64 -> transposed bf16 LDS [n][k]
#pragma unroll
    for (int i = 0; i < 8; ++i) {
      int idx = t + i * 256;                 // 2048 elements
      int k = idx >> 6, n = idx & 63;
      ldsWt[n][k] = (__bf16)W[(size_t)(kb + k) * N + (nbase + n)];
    }
    __syncthreads();

    const __bf16* arow = &ldsA[w * 16 + l16][0];
    v16bf af = make_afrag(arow, half);
#pragma unroll
    for (int nt = 0; nt < 4; ++nt) {
      v16bf bf_ = *(const v16bf*)&ldsWt[nt * 16 + l16][half * 16];
      acc[nt] = wmma_bf16(af, bf_, acc[nt]);
    }
    __syncthreads();
  }

  // store C: vgpr r -> row (half*8 + r), lane%16 -> col within 16-tile
#pragma unroll
  for (int nt = 0; nt < 4; ++nt) {
#pragma unroll
    for (int r = 0; r < 8; ++r) {
      int row = mbase + w * 16 + half * 8 + r;
      int col = nbase + nt * 16 + l16;
      size_t g = (size_t)row * N + col;
      if (C_IS_F32) ((float*)Cp)[g] = acc[nt][r];
      else          ((__bf16*)Cp)[g] = (__bf16)acc[nt][r];
    }
  }
}

// ---------------------------------------------------------------------------
// RoPE applied in-place to bf16 buffer laid out [B, S, nHeads, 64].
// One thread per (row, pair j<32).
// ---------------------------------------------------------------------------
__global__ __launch_bounds__(256)
void rope_kernel(__bf16* __restrict__ buf, const float* __restrict__ freqs,
                 int nHeads, int total) {
  int idx = blockIdx.x * blockDim.x + threadIdx.x;
  if (idx >= total) return;
  int j  = idx & 31;
  int rh = idx >> 5;                      // (b*S + s)*nHeads + h
  int s  = (rh / nHeads) % SS;
  size_t base = (size_t)rh * DH;
  float f = freqs[s * 32 + j];
  float c = __cosf(f), sn = __sinf(f);
  float x1 = (float)buf[base + j];
  float x2 = (float)buf[base + 32 + j];
  buf[base + j]      = (__bf16)(x1 * c - x2 * sn);
  buf[base + 32 + j] = (__bf16)(x2 * c + x1 * sn);
}

// ---------------------------------------------------------------------------
// Flash attention: grid (qtiles=16, H=9, B=4), block 256 (8 waves x 16 q rows).
// Q tile (128x64) and K blocks (64x64) staged by the Tensor Data Mover;
// V block transposed manually (TDM cannot transpose). Online softmax; all
// matmuls via WMMA bf16. ctx written bf16 [B,S,H,64].
// ---------------------------------------------------------------------------
__global__ __launch_bounds__(256)
void attn_kernel(const __bf16* __restrict__ qbuf, const __bf16* __restrict__ kbuf,
                 const __bf16* __restrict__ vbuf, __bf16* __restrict__ ctx) {
  __shared__ __attribute__((aligned(64))) __bf16 ldsQ[128][64];
  __shared__ __attribute__((aligned(64))) __bf16 ldsK[64][64];   // [key][d]
  __shared__ __attribute__((aligned(64))) __bf16 ldsVt[64][64];  // [d][key]
  __shared__ __attribute__((aligned(64))) __bf16 ldsP[128][64];  // probs, per-wave rows

  const int t    = threadIdx.x;
  const int lane = t & 31;
  const int w    = t >> 5;
  const int l16  = lane & 15;
  const int half = lane >> 4;

  const int qb  = blockIdx.x;       // 0..15
  const int h   = blockIdx.y;       // 0..8
  const int b   = blockIdx.z;       // 0..3
  const int kvh = h / (NH / NKV);   // GQA: 3 q heads share one kv head

  // TDM: load Q tile (128 rows x 64 d, source row stride H*D) into LDS
  if (w == 0) {
    tdm_load_2d((unsigned)(size_t)&ldsQ[0][0],
                qbuf + (((size_t)(b * SS + qb * 128) * NH) + h) * DH,
                /*tileW=*/DH, /*tileH=*/128, /*stride=*/NH * DH);
    tdm_wait0();
  }
  __syncthreads();

  float mrow[8], lrow[8];
#pragma unroll
  for (int r = 0; r < 8; ++r) { mrow[r] = -1e30f; lrow[r] = 0.f; }
  v8f o[4] = {vzero8(), vzero8(), vzero8(), vzero8()};

  const float scale = 0.125f;       // 1/sqrt(64)

  for (int kb = 0; kb < SS / 64; ++kb) {
    const size_t kvbase = (((size_t)(b * SS + kb * 64) * NKV) + kvh) * DH;

    // TDM: K block [64 keys][64 d] -> LDS (row stride KV*D)
    if (w == 0)
      tdm_load_2d((unsigned)(size_t)&ldsK[0][0], kbuf + kvbase,
                  /*tileW=*/DH, /*tileH=*/64, /*stride=*/NKV * DH);

    // manual transposed load of V block -> ldsVt[d][key]
#pragma unroll
    for (int i = 0; i < 16; ++i) {
      int idx = t + i * 256;        // 4096 elements
      int key = idx >> 6, d = idx & 63;
      ldsVt[d][key] = vbuf[kvbase + (size_t)key * NKV * DH / 64 * 64 + d];
    }
    if (w == 0) tdm_wait0();
    __syncthreads();

    // scores: S = Q @ K^T   (16 rows x 64 keys per wave)
    v8f sacc[4] = {vzero8(), vzero8(), vzero8(), vzero8()};
#pragma unroll
    for (int ks = 0; ks < 2; ++ks) {
      v16bf af = make_afrag(&ldsQ[w * 16 + l16][ks * 32], half);
#pragma unroll
      for (int nt = 0; nt < 4; ++nt) {
        v16bf bf_ = *(const v16bf*)&ldsK[nt * 16 + l16][ks * 32 + half * 16];
        sacc[nt] = wmma_bf16(af, bf_, sacc[nt]);
      }
    }

    // online softmax per row (rows half*8+r); lane reductions within 16-lane halves
#pragma unroll
    for (int r = 0; r < 8; ++r) {
      float s4[4];
      float rmax = -1e30f;
#pragma unroll
      for (int nt = 0; nt < 4; ++nt) {
        s4[nt] = sacc[nt][r] * scale;
        rmax = fmaxf(rmax, s4[nt]);
      }
#pragma unroll
      for (int m = 1; m < 16; m <<= 1)
        rmax = fmaxf(rmax, __shfl_xor(rmax, m, 32));

      float newm = fmaxf(mrow[r], rmax);
      float corr = __expf(mrow[r] - newm);
      mrow[r] = newm;

      int prow = w * 16 + half * 8 + r;
      float rsum = 0.f;
#pragma unroll
      for (int nt = 0; nt < 4; ++nt) {
        float p = __expf(s4[nt] - newm);
        rsum += p;
        ldsP[prow][nt * 16 + l16] = (__bf16)p;   // same-wave producer/consumer
      }
#pragma unroll
      for (int m = 1; m < 16; m <<= 1)
        rsum += __shfl_xor(rsum, m, 32);

      lrow[r] = lrow[r] * corr + rsum;
#pragma unroll
      for (int nt = 0; nt < 4; ++nt) o[nt][r] *= corr;
    }

    // O += P @ V   (P: 16x64 keys, V: 64 keys x 64 d)
#pragma unroll
    for (int ks = 0; ks < 2; ++ks) {
      v16bf af = make_afrag(&ldsP[w * 16 + l16][ks * 32], half);
#pragma unroll
      for (int nt = 0; nt < 4; ++nt) {
        v16bf bf_ = *(const v16bf*)&ldsVt[nt * 16 + l16][ks * 32 + half * 16];
        o[nt] = wmma_bf16(af, bf_, o[nt]);
      }
    }
    __syncthreads();   // before overwriting ldsK/ldsVt next iteration
  }

  // normalize and store ctx (bf16)
#pragma unroll
  for (int r = 0; r < 8; ++r) {
    float inv = 1.f / lrow[r];
    int row = qb * 128 + w * 16 + half * 8 + r;
#pragma unroll
    for (int nt = 0; nt < 4; ++nt) {
      int d = nt * 16 + l16;
      ctx[(((size_t)(b * SS + row) * NH) + h) * DH + d] = (__bf16)(o[nt][r] * inv);
    }
  }
}

// ---------------------------------------------------------------------------
// Host-side orchestration
// ---------------------------------------------------------------------------
extern "C" void kernel_launch(void* const* d_in, const int* in_sizes, int n_in,
                              void* d_out, int out_size, void* d_ws, size_t ws_size,
                              hipStream_t stream) {
  const float* hs    = (const float*)d_in[0];  // [B,S,HID]
  const float* freqs = (const float*)d_in[1];  // [2048,32]
  const float* Wq    = (const float*)d_in[2];  // [576,576]
  const float* Wk    = (const float*)d_in[3];  // [576,192]
  const float* Wv    = (const float*)d_in[4];  // [576,192]
  const float* Wo    = (const float*)d_in[5];  // [576,576]
  float* out = (float*)d_out;

  const int M  = BB * SS;           // 8192
  const size_t QE = (size_t)M * NH * DH;   // 4,718,592
  const size_t KE = (size_t)M * NKV * DH;  // 1,572,864

  char* ws = (char*)d_ws;
  __bf16* qbuf = (__bf16*)ws;                      ws += QE * sizeof(__bf16);
  __bf16* kbuf = (__bf16*)ws;                      ws += KE * sizeof(__bf16);
  __bf16* vbuf = (__bf16*)ws;                      ws += KE * sizeof(__bf16);
  __bf16* ctxb = (__bf16*)ws;                      // QE elements

  // 1) QKV projections (f32 -> bf16)
  wmma_gemm<true, false><<<dim3(M / 128, (NH * DH) / 64), 256, 0, stream>>>(
      hs, Wq, qbuf, M, NH * DH, HID);
  wmma_gemm<true, false><<<dim3(M / 128, (NKV * DH) / 64), 256, 0, stream>>>(
      hs, Wk, kbuf, M, NKV * DH, HID);
  wmma_gemm<true, false><<<dim3(M / 128, (NKV * DH) / 64), 256, 0, stream>>>(
      hs, Wv, vbuf, M, NKV * DH, HID);

  // 2) RoPE on Q and K
  {
    int qtot = M * NH * 32;
    int ktot = M * NKV * 32;
    rope_kernel<<<(qtot + 255) / 256, 256, 0, stream>>>(qbuf, freqs, NH, qtot);
    rope_kernel<<<(ktot + 255) / 256, 256, 0, stream>>>(kbuf, freqs, NKV, ktot);
  }

  // 3) attention (TDM-fed)
  attn_kernel<<<dim3(SS / 128, NH, BB), 256, 0, stream>>>(qbuf, kbuf, vbuf, ctxb);

  // 4) output projection (bf16 -> f32)
  wmma_gemm<false, true><<<dim3(M / 128, HID / 64), 256, 0, stream>>>(
      ctxb, Wo, out, M, HID, HID);
}
